// RGCN_30116310679598
// MI455X (gfx1250) — compile-verified, matrix operand order checked by reading, not run on hardware
//
#include <hip/hip_runtime.h>

// ---------------- problem constants ----------------
constexpr int NUMC    = 125;
constexpr int CH      = 512;
constexpr int VF      = 256;
constexpr int N_TGT   = 2048;
constexpr int N_MEM   = 40000;
constexpr int N_MAP   = 16;
constexpr int N_DDETR = 2048;
constexpr int N_TOTAL = N_TGT + N_MEM + N_MAP + N_DDETR;  // 44112
constexpr int E_MEM   = 640000;
constexpr int E_SG    = 131072;
constexpr int E_BR    = 400000;
constexpr int OFF_MEM = N_TGT;
constexpr int OFF_MAP = N_TGT + N_MEM;
constexpr int OFF_SG  = N_TGT + N_MEM + N_MAP;

typedef __attribute__((ext_vector_type(16))) __bf16 v16bf;
typedef __attribute__((ext_vector_type(8)))  __bf16 v8bf;
typedef __attribute__((ext_vector_type(4)))  __bf16 v4bf;
typedef __attribute__((ext_vector_type(8)))  float  v8f;
typedef __attribute__((ext_vector_type(4)))  unsigned int u32x4;
typedef __attribute__((ext_vector_type(8)))  int          i32x8;
typedef __attribute__((ext_vector_type(4)))  int          i32x4;

enum { GF_BIAS = 1, GF_ACCUM = 2, GF_RELU = 4 };

#define BM 128
#define BN 128
#define BK 32
#define LDSPITCH 40   // 32 bf16 data + 8 bf16 (16B) pad per row

#if defined(__has_builtin)
#  if __has_builtin(__builtin_amdgcn_tensor_load_to_lds) && __has_builtin(__builtin_amdgcn_s_wait_tensorcnt)
#    define HAVE_TDM 1
#  endif
#endif
#ifndef HAVE_TDM
#  define HAVE_TDM 0
#endif

__device__ __forceinline__ unsigned lds_off(const void* p) {
    // addrspace(3) -> generic keeps LDS byte offset in the low 32 bits
    return (unsigned)(unsigned long long)(uintptr_t)p;
}

#if HAVE_TDM
// Issue one TDM 2D tile load: tile = 128 rows x 32 bf16 (64B) from a row-major
// bf16 matrix (row stride = stride_elems), into LDS with 16B padding after
// every 64B row (row pitch becomes 80B = LDSPITCH bf16).
// rows_avail clips the tile against the matrix edge: OOB rows return zero.
__device__ __forceinline__ void tdm_tile_load(unsigned lds_addr, const __bf16* gsrc,
                                              int rows_avail, int stride_elems)
{
    unsigned long long ga = (unsigned long long)(uintptr_t)gsrc;
    u32x4 g0;
    g0[0] = 1u;                                             // count=1, user descriptor
    g0[1] = lds_addr;                                       // lds_addr (bytes)
    g0[2] = (unsigned)(ga & 0xFFFFFFFFull);                 // global_addr[31:0]
    g0[3] = (unsigned)((ga >> 32) & 0x01FFFFFFull)          // global_addr[56:32]
          | 0x80000000u;                                    // type=2 ("image")
    unsigned long long td0 = 0x7FFFFFFFull;                 // tensor_dim0: no K clip
    unsigned long long td1 = (unsigned long long)(unsigned)rows_avail; // tensor_dim1
    unsigned long long st0 = (unsigned long long)(unsigned)stride_elems;
    i32x8 g1;
    g1[0] = (int)((1u << 16)      // data_size = 2 bytes
                | (1u << 20)      // pad_enable
                | (3u << 22)      // pad_interval: every 16 DWORDs (64B)
                | (3u << 25));    // pad_amount: 4 DWORDs (16B)
    g1[1] = (int)((td0 & 0xFFFFull) << 16);                            // [63:48]=td0 lo16
    g1[2] = (int)(((td0 >> 16) & 0xFFFFull) | ((td1 & 0xFFFFull) << 16)); // td0 hi16 | td1 lo16
    g1[3] = (int)(((td1 >> 16) & 0xFFFFull) | ((unsigned)BK << 16));   // td1 hi16 | tile_dim0=32
    g1[4] = (int)(unsigned)BM;                                         // tile_dim1=128, tile_dim2=0
    g1[5] = (int)(st0 & 0xFFFFFFFFull);                                // dim0 stride lo32
    g1[6] = (int)((st0 >> 32) & 0xFFFFull);                            // dim0 stride hi16
    g1[7] = 0;
    i32x4 gz4 = {0, 0, 0, 0};
    i32x8 gz8 = {0, 0, 0, 0, 0, 0, 0, 0};
    __builtin_amdgcn_tensor_load_to_lds(g0, g1, gz4, gz4, gz8, 0);
}
#endif

// =====================================================================
// bf16 WMMA GEMM:  C[M,N] (f32) (+)= A[M,K]bf16 * Bt[N,K]bf16^T
// 256 threads (8 wave32), 128x128 tile, K-step 32, double-buffered LDS
// staged by the Tensor Data Mover; 8x v_wmma_f32_16x16x32_bf16 per step.
// =====================================================================
__launch_bounds__(256)
__global__ void gemm_bf16_wmma(const __bf16* __restrict__ A, int lda,
                               const __bf16* __restrict__ Bt, int ldb,
                               float* __restrict__ Cm, int ldc,
                               const float* __restrict__ bias,
                               int M, int N, int K, int flags)
{
    __shared__ __attribute__((aligned(16))) __bf16 sA[2][BM][LDSPITCH];
    __shared__ __attribute__((aligned(16))) __bf16 sB[2][BN][LDSPITCH];

    const int tid  = threadIdx.x;
    const int lane = tid & 31;
    const int wave = tid >> 5;
    const int wm   = wave & 3;
    const int wn   = wave >> 2;
    const int lm   = lane & 15;
    const int lh   = lane >> 4;
    const int m0   = blockIdx.x * BM;
    const int n0   = blockIdx.y * BN;

    v8f acc[2][4];
    #pragma unroll
    for (int i = 0; i < 2; ++i)
        #pragma unroll
        for (int j = 0; j < 4; ++j)
            #pragma unroll
            for (int q = 0; q < 8; ++q) acc[i][j][q] = 0.0f;

#if HAVE_TDM
    const bool isw0 = ((__builtin_amdgcn_readfirstlane((int)threadIdx.x) >> 5) == 0);
    if (isw0) {
        tdm_tile_load(lds_off(&sA[0][0][0]), A + (size_t)m0 * lda, M - m0, lda);
        tdm_tile_load(lds_off(&sB[0][0][0]), Bt + (size_t)n0 * ldb, N - n0, ldb);
    }
#endif

    int cur = 0;
    for (int k0 = 0; k0 < K; k0 += BK) {
#if HAVE_TDM
        if (isw0) __builtin_amdgcn_s_wait_tensorcnt(0);  // current tiles landed
        __syncthreads();                                 // visible to all; prev readers done
        if (isw0 && (k0 + BK) < K) {                     // overlap next DMA with compute
            tdm_tile_load(lds_off(&sA[cur ^ 1][0][0]), A + (size_t)m0 * lda + k0 + BK,
                          M - m0, lda);
            tdm_tile_load(lds_off(&sB[cur ^ 1][0][0]), Bt + (size_t)n0 * ldb + k0 + BK,
                          N - n0, ldb);
        }
#else
        // fallback: register-staged bf16 tiles
        #pragma unroll
        for (int it = 0; it < 2; ++it) {
            int idx = it * 256 + tid;              // 512 8-elem slots
            int r = idx >> 2, c8 = (idx & 3) << 3;
            v8bf v = {};
            int gm = m0 + r;
            if (gm < M && (k0 + c8) < K)
                v = *(const v8bf*)(A + (size_t)gm * lda + k0 + c8);
            *(v8bf*)&sA[cur][r][c8] = v;
            v8bf w = {};
            int gn = n0 + r;
            if (gn < N && (k0 + c8) < K)
                w = *(const v8bf*)(Bt + (size_t)gn * ldb + k0 + c8);
            *(v8bf*)&sB[cur][r][c8] = w;
        }
        __syncthreads();
#endif
        // fragments per ISA 16-bit 16x32 layout: lane-half lh holds
        // K = lh*8..+7 (elems 0-7) and 16+lh*8..+7 (elems 8-15)
        v16bf af[2], bfr[4];
        #pragma unroll
        for (int tm = 0; tm < 2; ++tm) {
            const __bf16* p = &sA[cur][wm * 32 + tm * 16 + lm][lh * 8];
            v8bf lo = *(const v8bf*)p;
            v8bf hi = *(const v8bf*)(p + 16);
            af[tm] = __builtin_shufflevector(lo, hi, 0,1,2,3,4,5,6,7,8,9,10,11,12,13,14,15);
        }
        #pragma unroll
        for (int tn = 0; tn < 4; ++tn) {
            const __bf16* p = &sB[cur][wn * 64 + tn * 16 + lm][lh * 8];
            v8bf lo = *(const v8bf*)p;
            v8bf hi = *(const v8bf*)(p + 16);
            bfr[tn] = __builtin_shufflevector(lo, hi, 0,1,2,3,4,5,6,7,8,9,10,11,12,13,14,15);
        }
        #pragma unroll
        for (int tm = 0; tm < 2; ++tm)
            #pragma unroll
            for (int tn = 0; tn < 4; ++tn)
                acc[tm][tn] = __builtin_amdgcn_wmma_f32_16x16x32_bf16(
                    false, af[tm], false, bfr[tn], (short)0, acc[tm][tn], false, false);
#if !HAVE_TDM
        __syncthreads();
#endif
        cur ^= 1;
    }
#if HAVE_TDM
    __syncthreads();
#endif

    // epilogue: C/D layout VGPR i -> M=i (lanes 0-15) / i+8 (lanes 16-31), N=lane%16
    #pragma unroll
    for (int tm = 0; tm < 2; ++tm)
        #pragma unroll
        for (int tn = 0; tn < 4; ++tn)
            #pragma unroll
            for (int i = 0; i < 8; ++i) {
                int m = m0 + wm * 32 + tm * 16 + i + lh * 8;
                int n = n0 + wn * 64 + tn * 16 + lm;
                if (m < M && n < N) {
                    float v = acc[tm][tn][i];
                    if (flags & GF_BIAS) v += bias[n];
                    float* cp = Cm + (size_t)m * ldc + n;
                    if (flags & GF_ACCUM) v += *cp;
                    if (flags & GF_RELU) v = fmaxf(v, 0.f);
                    *cp = v;
                }
            }
}

// =====================================================================
// conversion / graph kernels
// =====================================================================
__global__ void cvt_bf16(__bf16* __restrict__ dst, const float* __restrict__ src,
                         long long n4) {                   // n4 = n/4 float4 chunks
    long long t = blockIdx.x * (long long)blockDim.x + threadIdx.x;
    if (t >= n4) return;
    float4 v = *(const float4*)(src + t * 4);
    v4bf b; b[0] = (__bf16)v.x; b[1] = (__bf16)v.y; b[2] = (__bf16)v.z; b[3] = (__bf16)v.w;
    *(v4bf*)(dst + t * 4) = b;
}
// batched transpose-convert: src = B mats of [K][N] f32 -> dst [b][N][K] bf16
__global__ void cvtT_bf16(__bf16* __restrict__ dst, const float* __restrict__ src,
                          int Bc, int K, int N) {
    long long t = blockIdx.x * (long long)blockDim.x + threadIdx.x;
    long long per = (long long)K * N;
    if (t >= per * Bc) return;
    int b = (int)(t / per);
    long long r = t - (long long)b * per;
    int n = (int)(r / K), k = (int)(r - (long long)n * K);
    dst[(long long)b * per + (long long)n * K + k] =
        (__bf16)src[(long long)b * per + (long long)k * N + n];
}
__global__ void fill_f32(float* p, float v, long long n) {
    long long t = blockIdx.x * (long long)blockDim.x + threadIdx.x;
    if (t < n) p[t] = v;
}
__global__ void fill_i32(int* p, int v, int n) {
    int t = blockIdx.x * blockDim.x + threadIdx.x;
    if (t < n) p[t] = v;
}
__global__ void gather_rows(float* __restrict__ dst, int ldd, int col_off,
                            const float* __restrict__ emb, int width,
                            const int* __restrict__ ids, int n) {
    long long t = blockIdx.x * (long long)blockDim.x + threadIdx.x;
    if (t >= (long long)n * width) return;
    int i = (int)(t / width), c = (int)(t % width);
    dst[(size_t)i * ldd + col_off + c] = emb[(size_t)ids[i] * width + c];
}
__global__ void count_edges(int* __restrict__ cnt, const int* __restrict__ dst,
                            const int* __restrict__ et, int r, int E) {
    int e = blockIdx.x * blockDim.x + threadIdx.x;
    if (e < E && et[e] == r) atomicAdd(&cnt[dst[e]], 1);
}
__global__ void scatter_rel(float* __restrict__ agg, const float* __restrict__ x,
                            const int* __restrict__ src, const int* __restrict__ dst,
                            const int* __restrict__ et, int r, int E) {
    long long t = blockIdx.x * (long long)blockDim.x + threadIdx.x;
    if (t >= (long long)E * (CH / 4)) return;
    int e = (int)(t >> 7);
    int c = (int)(t & 127) << 2;
    if (et[e] != r) return;
    float4 v = *(const float4*)(x + (size_t)src[e] * CH + c);
    float* a = agg + (size_t)dst[e] * CH + c;
    atomicAdd(a + 0, v.x); atomicAdd(a + 1, v.y);
    atomicAdd(a + 2, v.z); atomicAdd(a + 3, v.w);
}
// mean-normalize rows AND convert to bf16 (feeds WMMA GEMM directly)
__global__ void scale_cvt(__bf16* __restrict__ dst, const float* __restrict__ agg,
                          const int* __restrict__ cnt, int Nn) {
    long long t = blockIdx.x * (long long)blockDim.x + threadIdx.x;
    if (t >= (long long)Nn * (CH / 4)) return;
    int i = (int)(t >> 7);
    float s = 1.0f / fmaxf((float)cnt[i], 1.0f);
    float4 v = *(const float4*)(agg + t * 4);
    v4bf b; b[0] = (__bf16)(v.x * s); b[1] = (__bf16)(v.y * s);
    b[2] = (__bf16)(v.z * s); b[3] = (__bf16)(v.w * s);
    *(v4bf*)(dst + t * 4) = b;
}
__global__ void deg_count(float* __restrict__ deg, const int* __restrict__ dst, int E) {
    int e = blockIdx.x * blockDim.x + threadIdx.x;
    if (e < E) atomicAdd(&deg[dst[e]], 1.0f);
}
__global__ void rsqrt_inplace(float* deg, int n) {
    int t = blockIdx.x * blockDim.x + threadIdx.x;
    if (t < n) deg[t] = rsqrtf(fmaxf(deg[t], 1.0f));
}
__global__ void scatter_gcn(float* __restrict__ agg, const float* __restrict__ x,
                            const float* __restrict__ dis,
                            const int* __restrict__ src, const int* __restrict__ dst, int E) {
    long long t = blockIdx.x * (long long)blockDim.x + threadIdx.x;
    if (t >= (long long)E * (CH / 4)) return;
    int e = (int)(t >> 7);
    int c = (int)(t & 127) << 2;
    int s = src[e], d = dst[e];
    float w = dis[s] * dis[d];
    float4 v = *(const float4*)(x + (size_t)s * CH + c);
    float* a = agg + (size_t)d * CH + c;
    atomicAdd(a + 0, v.x * w); atomicAdd(a + 1, v.y * w);
    atomicAdd(a + 2, v.z * w); atomicAdd(a + 3, v.w * w);
}
// self-loop add AND convert to bf16
__global__ void self_loop_cvt(__bf16* __restrict__ dst, const float* __restrict__ agg,
                              const float* __restrict__ x, const float* __restrict__ dis,
                              int Nn) {
    long long t = blockIdx.x * (long long)blockDim.x + threadIdx.x;
    if (t >= (long long)Nn * (CH / 4)) return;
    int i = (int)(t >> 7);
    float w = dis[i] * dis[i];
    float4 a = *(const float4*)(agg + t * 4);
    float4 v = *(const float4*)(x + t * 4);
    v4bf b;
    b[0] = (__bf16)(a.x + v.x * w); b[1] = (__bf16)(a.y + v.y * w);
    b[2] = (__bf16)(a.z + v.z * w); b[3] = (__bf16)(a.w + v.w * w);
    *(v4bf*)(dst + t * 4) = b;
}

// =====================================================================
// host-side orchestration
// =====================================================================
static inline int cdiv(long long a, long long b) { return (int)((a + b - 1) / b); }

static void launch_gemm(const __bf16* A, int lda, const __bf16* Bt, int ldb,
                        float* Cm, int ldc, const float* bias,
                        int M, int N, int K, int flags, hipStream_t s) {
    dim3 g(cdiv(M, BM), cdiv(N, BN));
    gemm_bf16_wmma<<<g, 256, 0, s>>>(A, lda, Bt, ldb, Cm, ldc, bias, M, N, K, flags);
}

// out = relu( sum_r mean_r(A_r x) W_r + x root + b );  Wt/rootT are bf16 [N][K]
static void rgcn_layer(const float* x, float* out, float* agg, __bf16* abf, int* cnt,
                       int Nn, int R, const __bf16* Wt, const __bf16* rootT,
                       const float* b, const int* src, const int* dst, const int* et,
                       int E, hipStream_t s) {
    for (int r = 0; r < R; ++r) {
        fill_i32<<<cdiv(Nn, 256), 256, 0, s>>>(cnt, 0, Nn);
        count_edges<<<cdiv(E, 256), 256, 0, s>>>(cnt, dst, et, r, E);
        fill_f32<<<cdiv((long long)Nn * CH, 256), 256, 0, s>>>(agg, 0.f, (long long)Nn * CH);
        scatter_rel<<<cdiv((long long)E * (CH / 4), 256), 256, 0, s>>>(agg, x, src, dst, et, r, E);
        scale_cvt<<<cdiv((long long)Nn * (CH / 4), 256), 256, 0, s>>>(abf, agg, cnt, Nn);
        launch_gemm(abf, CH, Wt + (size_t)r * CH * CH, CH, out, CH, b,
                    Nn, CH, CH, (r == 0 ? GF_BIAS : GF_ACCUM), s);
    }
    cvt_bf16<<<cdiv((long long)Nn * (CH / 4), 256), 256, 0, s>>>(abf, x, (long long)Nn * (CH / 4));
    launch_gemm(abf, CH, rootT, CH, out, CH, nullptr, Nn, CH, CH, GF_ACCUM | GF_RELU, s);
}

static void gcn_layer(const float* x, float* out, float* agg, __bf16* abf, float* deg,
                      const __bf16* Wt, const float* b,
                      const int* src, const int* dst, int E, int Nn, hipStream_t s) {
    fill_f32<<<cdiv(Nn, 256), 256, 0, s>>>(deg, 1.0f, Nn);
    deg_count<<<cdiv(E, 256), 256, 0, s>>>(deg, dst, E);
    rsqrt_inplace<<<cdiv(Nn, 256), 256, 0, s>>>(deg, Nn);
    fill_f32<<<cdiv((long long)Nn * CH, 256), 256, 0, s>>>(agg, 0.f, (long long)Nn * CH);
    scatter_gcn<<<cdiv((long long)E * (CH / 4), 256), 256, 0, s>>>(agg, x, deg, src, dst, E);
    self_loop_cvt<<<cdiv((long long)Nn * (CH / 4), 256), 256, 0, s>>>(abf, agg, x, deg, Nn);
    launch_gemm(abf, CH, Wt, CH, out, CH, b, Nn, CH, CH, GF_BIAS | GF_RELU, s);
}

extern "C" void kernel_launch(void* const* d_in, const int* in_sizes, int n_in,
                              void* d_out, int out_size, void* d_ws, size_t ws_size,
                              hipStream_t stream) {
    const int*   obj_ids_tgt   = (const int*)  d_in[0];
    const float* obj_feats_tgt = (const float*)d_in[1];
    const int*   obj_ids_mem   = (const int*)  d_in[2];
    const float* obj_feats_mem = (const float*)d_in[3];
    const int*   ei_mem        = (const int*)  d_in[4];
    const int*   et_mem        = (const int*)  d_in[5];
    const int*   map_ids       = (const int*)  d_in[6];
    const int*   ddetr_labels  = (const int*)  d_in[7];
    const float* feat_ddetr    = (const float*)d_in[8];
    const int*   ei_sg         = (const int*)  d_in[9];
    const int*   ei_br         = (const int*)  d_in[10];
    const int*   et_br         = (const int*)  d_in[11];
    const float* obj_emb   = (const float*)d_in[12];
    const float* scene_emb = (const float*)d_in[13];
    const float* proj_w    = (const float*)d_in[14];
    const float* proj_b    = (const float*)d_in[15];
    const float* mem_W     = (const float*)d_in[16];
    const float* mem_root  = (const float*)d_in[17];
    const float* mem_b     = (const float*)d_in[18];
    const float* sg_W      = (const float*)d_in[19];
    const float* sg_b      = (const float*)d_in[20];
    const float* br_W      = (const float*)d_in[21];
    const float* br_root   = (const float*)d_in[22];
    const float* br_b      = (const float*)d_in[23];
    const float* pred_w1   = (const float*)d_in[24];
    const float* pred_b1   = (const float*)d_in[25];
    const float* pred_w2   = (const float*)d_in[26];
    const float* pred_b2   = (const float*)d_in[27];

    // ---- workspace layout (16B aligned sections) ----
    const size_t NT = (size_t)N_TOTAL * CH;                 // 22.6M elems
    const size_t MAT = (size_t)CH * CH;                     // 262144
    float*  bufA = (float*)d_ws;
    float*  bufB = bufA + NT;
    float*  bufS = bufB + NT;
    __bf16* abf  = (__bf16*)(bufS + NT);                    // bf16 A-operand scratch
    __bf16* wbf  = abf + NT;                                // transposed bf16 weights
    const size_t WO_PROJ = 0;
    const size_t WO_MEMW = WO_PROJ + (size_t)VF * VF;
    const size_t WO_MEMR = WO_MEMW + 24 * MAT;
    const size_t WO_SGW  = WO_MEMR + 3 * MAT;
    const size_t WO_BRW  = WO_SGW + 3 * MAT;
    const size_t WO_BRR  = WO_BRW + 16 * MAT;
    const size_t WO_PW1  = WO_BRR + 4 * MAT;
    const size_t WO_PW2  = WO_PW1 + MAT;
    const size_t WTOTAL  = WO_PW2 + (size_t)NUMC * CH;
    size_t wpad = (WTOTAL + 7) & ~(size_t)7;
    int*   cnt = (int*)(wbf + wpad);
    float* deg = (float*)(cnt + N_TOTAL);

    hipStream_t s = stream;

    // ---- one-time per call: transpose+convert all weights to bf16 [N][K] ----
    cvtT_bf16<<<cdiv((long long)VF * VF, 256), 256, 0, s>>>(wbf + WO_PROJ, proj_w, 1, VF, VF);
    cvtT_bf16<<<cdiv(24LL * MAT, 256), 256, 0, s>>>(wbf + WO_MEMW, mem_W, 24, CH, CH);
    cvtT_bf16<<<cdiv(3LL * MAT, 256), 256, 0, s>>>(wbf + WO_MEMR, mem_root, 3, CH, CH);
    cvtT_bf16<<<cdiv(3LL * MAT, 256), 256, 0, s>>>(wbf + WO_SGW, sg_W, 3, CH, CH);
    cvtT_bf16<<<cdiv(16LL * MAT, 256), 256, 0, s>>>(wbf + WO_BRW, br_W, 16, CH, CH);
    cvtT_bf16<<<cdiv(4LL * MAT, 256), 256, 0, s>>>(wbf + WO_BRR, br_root, 4, CH, CH);
    cvtT_bf16<<<cdiv(1LL * MAT, 256), 256, 0, s>>>(wbf + WO_PW1, pred_w1, 1, CH, CH);
    cvtT_bf16<<<cdiv((long long)CH * NUMC, 256), 256, 0, s>>>(wbf + WO_PW2, pred_w2, 1, CH, NUMC);

    // ---- node feature construction into bufA ----
    gather_rows<<<cdiv((long long)N_TGT * VF, 256), 256, 0, s>>>(
        bufA, CH, 0, obj_emb, VF, obj_ids_tgt, N_TGT);
    cvt_bf16<<<cdiv((long long)N_TGT * VF / 4, 256), 256, 0, s>>>(abf, obj_feats_tgt,
                                                                  (long long)N_TGT * VF / 4);
    launch_gemm(abf, VF, wbf + WO_PROJ, VF, bufA + VF, CH, proj_b, N_TGT, VF, VF, GF_BIAS, s);

    gather_rows<<<cdiv((long long)N_MEM * VF, 256), 256, 0, s>>>(
        bufA + (size_t)OFF_MEM * CH, CH, 0, obj_emb, VF, obj_ids_mem, N_MEM);
    cvt_bf16<<<cdiv((long long)N_MEM * VF / 4, 256), 256, 0, s>>>(abf, obj_feats_mem,
                                                                  (long long)N_MEM * VF / 4);
    launch_gemm(abf, VF, wbf + WO_PROJ, VF, bufA + (size_t)OFF_MEM * CH + VF, CH, proj_b,
                N_MEM, VF, VF, GF_BIAS, s);

    gather_rows<<<cdiv((long long)N_MAP * CH, 256), 256, 0, s>>>(
        bufA + (size_t)OFF_MAP * CH, CH, 0, scene_emb, CH, map_ids, N_MAP);

    cvt_bf16<<<cdiv((long long)N_DDETR * VF / 4, 256), 256, 0, s>>>(abf, feat_ddetr,
                                                                    (long long)N_DDETR * VF / 4);
    launch_gemm(abf, VF, wbf + WO_PROJ, VF, bufA + (size_t)OFF_SG * CH, CH, proj_b,
                N_DDETR, VF, VF, GF_BIAS, s);
    gather_rows<<<cdiv((long long)N_DDETR * VF, 256), 256, 0, s>>>(
        bufA + (size_t)OFF_SG * CH, CH, VF, obj_emb, VF, ddetr_labels, N_DDETR);

    // ---- 3-layer RGCN on memory graph ----
    {
        float* xin  = bufA + (size_t)OFF_MEM * CH;
        float* xout = bufB + (size_t)OFF_MEM * CH;
        for (int i = 0; i < 3; ++i) {
            rgcn_layer(xin, xout, bufS, abf, cnt, N_MEM, 8,
                       wbf + WO_MEMW + (size_t)i * 8 * MAT,
                       wbf + WO_MEMR + (size_t)i * MAT,
                       mem_b + (size_t)i * CH,
                       ei_mem, ei_mem + E_MEM, et_mem, E_MEM, s);
            float* t = xin; xin = xout; xout = t;
        }
        (void)hipMemcpyAsync(bufA + (size_t)OFF_MEM * CH, bufB + (size_t)OFF_MEM * CH,
                             (size_t)N_MEM * CH * sizeof(float), hipMemcpyDeviceToDevice, s);
    }
    // ---- 3-layer GCN on scene graph ----
    {
        float* xin  = bufA + (size_t)OFF_SG * CH;
        float* xout = bufB + (size_t)OFF_SG * CH;
        for (int i = 0; i < 3; ++i) {
            gcn_layer(xin, xout, bufS, abf, deg,
                      wbf + WO_SGW + (size_t)i * MAT, sg_b + (size_t)i * CH,
                      ei_sg, ei_sg + E_SG, E_SG, N_DDETR, s);
            float* t = xin; xin = xout; xout = t;
        }
        (void)hipMemcpyAsync(bufA + (size_t)OFF_SG * CH, bufB + (size_t)OFF_SG * CH,
                             (size_t)N_DDETR * CH * sizeof(float), hipMemcpyDeviceToDevice, s);
    }
    // ---- 4-layer RGCN on bridge graph (ends back in bufA) ----
    {
        float* xin = bufA;
        float* xout = bufB;
        for (int i = 0; i < 4; ++i) {
            rgcn_layer(xin, xout, bufS, abf, cnt, N_TOTAL, 4,
                       wbf + WO_BRW + (size_t)i * 4 * MAT,
                       wbf + WO_BRR + (size_t)i * MAT,
                       br_b + (size_t)i * CH,
                       ei_br, ei_br + E_BR, et_br, E_BR, s);
            float* t = xin; xin = xout; xout = t;
        }
    }
    // ---- predictor on target nodes ----
    cvt_bf16<<<cdiv((long long)N_TGT * CH / 4, 256), 256, 0, s>>>(abf, bufA,
                                                                  (long long)N_TGT * CH / 4);
    launch_gemm(abf, CH, wbf + WO_PW1, CH, bufB, CH, pred_b1,
                N_TGT, CH, CH, GF_BIAS | GF_RELU, s);
    cvt_bf16<<<cdiv((long long)N_TGT * CH / 4, 256), 256, 0, s>>>(abf, bufB,
                                                                  (long long)N_TGT * CH / 4);
    launch_gemm(abf, CH, wbf + WO_PW2, CH, (float*)d_out, NUMC, pred_b2,
                N_TGT, NUMC, CH, GF_BIAS, s);
}
